// BinomialLoss_45569603010930
// MI455X (gfx1250) — matrix operand order repaired and checked by reading, not compiled, not run on hardware
//
#include <hip/hip_runtime.h>
#include <hip/hip_bf16.h>

typedef __attribute__((ext_vector_type(16))) _Float16 v16h;
typedef __attribute__((ext_vector_type(8)))  _Float16 v8h;
typedef __attribute__((ext_vector_type(8)))  float    v8f;
typedef __attribute__((ext_vector_type(4)))  int      v4i;
typedef __attribute__((address_space(1)))    v4i      g_v4i;   // global int4
typedef __attribute__((address_space(3)))    v4i      s_v4i;   // LDS int4

#define N_PTS   4096
#define D_DIM   512
#define ALPHA_C 40.0f
#define MARGIN_C 0.5f
#define COL_CHUNKS 4
#define TILES_PER_CHUNK ((N_PTS / 16) / COL_CHUNKS)   // 64 column tiles per chunk
#define B_STRIDE 520        // halves per LDS row: 512 + 8 pad (bank de-conflict)

#if __has_builtin(__builtin_amdgcn_global_load_async_to_lds_b128)
#define USE_ASYNC_LDS 1
#else
#define USE_ASYNC_LDS 0
#endif

// stable softplus: log(1+exp(z)) = max(z,0) + log1p(exp(-|z|))
__device__ __forceinline__ float sp(float z) {
    return fmaxf(z, 0.0f) + __logf(1.0f + __expf(-fabsf(z)));
}

__global__ void convert_f32_to_f16(const float* __restrict__ x,
                                   _Float16* __restrict__ xh, int n) {
    int i = blockIdx.x * blockDim.x + threadIdx.x;
    if (i < n) xh[i] = (_Float16)x[i];
}

// Stage one 16-row B tile (16 KB contiguous in global) into padded LDS rows.
// 256 threads x 4 b128 each = 1024 x 16B.
__device__ __forceinline__ void stage_tile(const _Float16* __restrict__ xh,
                                           _Float16* lds_buf, int jbase, int tid) {
    #pragma unroll
    for (int p = 0; p < 4; ++p) {
        const int idx = p * 256 + tid;          // 0..1023
        const int row = idx >> 6;               // 0..15
        const int c16 = idx & 63;               // 16B chunk within row
        const _Float16* g = xh + (size_t)(jbase + row) * D_DIM + c16 * 8;
        _Float16* l = lds_buf + row * B_STRIDE + c16 * 8;
#if USE_ASYNC_LDS
        __builtin_amdgcn_global_load_async_to_lds_b128((g_v4i*)g, (s_v4i*)l, 0, 0);
#else
        *(v4i*)l = *(const v4i*)g;
#endif
    }
}

__device__ __forceinline__ void wait_stage() {
#if USE_ASYNC_LDS
#if __has_builtin(__builtin_amdgcn_s_wait_asynccnt)
    __builtin_amdgcn_s_wait_asynccnt(0);
#else
    asm volatile("s_wait_asynccnt 0x0" ::: "memory");
#endif
#endif
    __syncthreads();
}

// One wave per (row-strip of 16, column chunk of 1024).
// grid = 128 blocks x 256 threads = 1024 waves = 256 strips x 4 chunks.
// All 8 waves of a block share the same column chunk -> B tiles staged in LDS.
__global__ __launch_bounds__(256)
void gemm_loss(const _Float16* __restrict__ xh, const int* __restrict__ tgt,
               float* __restrict__ rowPos, float* __restrict__ rowPosCnt,
               float* __restrict__ rowNeg, float* __restrict__ rowNegCnt,
               float* __restrict__ lastAcc /* [2*COL_CHUNKS] */) {
    __shared__ _Float16 bs[2][16 * B_STRIDE];   // 2 x 16.25 KB double buffer

    const int tid  = threadIdx.x;
    const int lane = tid & 31;
    const int wid  = tid >> 5;
    const int gw   = blockIdx.x * 8 + wid;      // 0..1023
    const int strip = gw & 255;                 // row strip 0..255
    const int chunk = gw >> 8;                  // 0..3  (uniform per block)
    const int row_base = strip * 16;
    const int hi  = lane >> 4;                  // 0 or 1 (half-wave)
    const int l16 = lane & 15;

    // ---- A fragments: 16 x (16x32 f16), strip resident in VGPRs -------------
    // ISA layout (16-bit A 16x32): lanes 0-15 row=lane, K {k0+0..7, k0+16..23};
    // lanes 16-31 row=lane-16, K {k0+8..15, k0+24..31}.
    v16h a[16];
    const _Float16* arow = xh + (size_t)(row_base + l16) * D_DIM;
    #pragma unroll
    for (int kk = 0; kk < 16; ++kk) {
        const int k0 = kk * 32 + hi * 8;
        v8h lo = *(const v8h*)(arow + k0);
        v8h hq = *(const v8h*)(arow + k0 + 16);
        a[kk] = __builtin_shufflevector(lo, hq, 0,1,2,3,4,5,6,7,8,9,10,11,12,13,14,15);
    }

    // C/D layout: VGPR v <-> row M = v + 8*hi; lane%16 <-> column N.
    int rowT[8];
    #pragma unroll
    for (int v = 0; v < 8; ++v) rowT[v] = tgt[row_base + v + 8 * hi];

    float ps[8], pc[8], ns[8], nc[8];
    #pragma unroll
    for (int v = 0; v < 8; ++v) { ps[v] = 0.f; pc[v] = 0.f; ns[v] = 0.f; nc[v] = 0.f; }
    float lp = 0.f, ln_ = 0.f;   // last-row (n-1) sim sums under pos/neg masks

    const int tile0 = chunk * TILES_PER_CHUNK;

    stage_tile(xh, bs[0], tile0 * 16, tid);
    wait_stage();

    for (int t = 0; t < TILES_PER_CHUNK; ++t) {
        const int jbase = (tile0 + t) * 16;
        if (t + 1 < TILES_PER_CHUNK)
            stage_tile(xh, bs[(t + 1) & 1], jbase + 16, tid);   // overlap with compute

        // B (32x16 f16) from LDS: lane holds column N=l16, K halves
        // {kk*32 + hi*8 + 0..7, kk*32 + 16 + hi*8 + 0..7} within the padded row.
        const _Float16* brow = &bs[t & 1][l16 * B_STRIDE + hi * 8];

        v8f c = {0.f, 0.f, 0.f, 0.f, 0.f, 0.f, 0.f, 0.f};
        #pragma unroll
        for (int kk = 0; kk < 16; ++kk) {
            v8h b0 = *(const v8h*)(brow + kk * 32);
            v8h b1 = *(const v8h*)(brow + kk * 32 + 16);
            v16h b = __builtin_shufflevector(b0, b1, 0,1,2,3,4,5,6,7,8,9,10,11,12,13,14,15);
            c = __builtin_amdgcn_wmma_f32_16x16x32_f16(
                    false, a[kk], false, b, (short)0, c, false, false);
        }

        const int tcol = tgt[jbase + l16];
        #pragma unroll
        for (int v = 0; v < 8; ++v) {
            const float sim = c[v];
            const int grow = row_base + v + 8 * hi;
            if (rowT[v] == tcol) {
                if (sim < 1.0f) {                      // pos pair (excludes self)
                    ps[v] += sp(-2.0f * (sim - MARGIN_C));
                    pc[v] += 1.0f;
                    if (grow == N_PTS - 1) lp += sim;
                }
            } else {                                   // neg pair
                ns[v] += sp(ALPHA_C * (sim - MARGIN_C));
                nc[v] += 1.0f;
                if (grow == N_PTS - 1) ln_ += sim;
            }
        }

        wait_stage();   // own async stage done + all waves done with both buffers
    }

    // ---- reduce across the 16 lanes sharing each row ------------------------
    #pragma unroll
    for (int v = 0; v < 8; ++v) {
        #pragma unroll
        for (int m = 1; m < 16; m <<= 1) {
            ps[v] += __shfl_xor(ps[v], m, 32);
            pc[v] += __shfl_xor(pc[v], m, 32);
            ns[v] += __shfl_xor(ns[v], m, 32);
            nc[v] += __shfl_xor(nc[v], m, 32);
        }
    }
    #pragma unroll
    for (int m = 1; m < 16; m <<= 1) {
        lp  += __shfl_xor(lp,  m, 32);
        ln_ += __shfl_xor(ln_, m, 32);
    }

    if (l16 == 0) {   // lane 0 -> rows 0..7 of strip, lane 16 -> rows 8..15
        #pragma unroll
        for (int v = 0; v < 8; ++v) {
            const int r = row_base + v + 8 * hi;
            const size_t idx = (size_t)chunk * N_PTS + r;
            rowPos[idx]    = ps[v];
            rowPosCnt[idx] = pc[v];
            rowNeg[idx]    = ns[v];
            rowNegCnt[idx] = nc[v];
        }
        if (strip == 255 && hi == 1) {   // owner of row N-1
            lastAcc[2 * chunk]     = lp;
            lastAcc[2 * chunk + 1] = ln_;
        }
    }
}

__global__ void finalize(const float* __restrict__ rowPos, const float* __restrict__ rowPosCnt,
                         const float* __restrict__ rowNeg, const float* __restrict__ rowNegCnt,
                         const float* __restrict__ lastAcc, float* __restrict__ out) {
    __shared__ float sLoss[256];
    __shared__ float sInv[256];
    float loss = 0.f, inv = 0.f;
    for (int r = threadIdx.x; r < N_PTS; r += 256) {
        float ps = 0.f, pc = 0.f, ns = 0.f, nc = 0.f;
        for (int c = 0; c < COL_CHUNKS; ++c) {
            const size_t i = (size_t)c * N_PTS + r;
            ps += rowPos[i]; pc += rowPosCnt[i];
            ns += rowNeg[i]; nc += rowNegCnt[i];
        }
        if (nc > 0.f) {
            loss += ps / fmaxf(pc, 1.f) + ns / fmaxf(nc, 1.f);
        } else {
            inv += 1.f;
        }
    }
    sLoss[threadIdx.x] = loss;
    sInv[threadIdx.x]  = inv;
    __syncthreads();
    for (int s = 128; s > 0; s >>= 1) {
        if (threadIdx.x < s) {
            sLoss[threadIdx.x] += sLoss[threadIdx.x + s];
            sInv[threadIdx.x]  += sInv[threadIdx.x + s];
        }
        __syncthreads();
    }
    if (threadIdx.x == 0) {
        out[0] = sLoss[0] / (float)N_PTS;                 // loss
        out[1] = sInv[0] / (float)N_PTS;                  // prec
        float lpS = 0.f, lnS = 0.f, pcL = 0.f, ncL = 0.f;
        for (int c = 0; c < COL_CHUNKS; ++c) {
            lpS += lastAcc[2 * c];
            lnS += lastAcc[2 * c + 1];
            const size_t i = (size_t)c * N_PTS + (N_PTS - 1);
            pcL += rowPosCnt[i];
            ncL += rowNegCnt[i];
        }
        out[2] = lpS / fmaxf(pcL, 1.f);                   // last_pos
        out[3] = lnS / fmaxf(ncL, 1.f);                   // last_neg
    }
}

extern "C" void kernel_launch(void* const* d_in, const int* in_sizes, int n_in,
                              void* d_out, int out_size, void* d_ws, size_t ws_size,
                              hipStream_t stream) {
    const float* x   = (const float*)d_in[0];
    const int*   tgt = (const int*)d_in[1];
    float*       out = (float*)d_out;

    char* ws = (char*)d_ws;
    // layout: [xh: N*D f16 = 4 MiB][4 row arrays: 4*COL_CHUNKS*N f32][lastAcc: 8 f32]
    _Float16* xh = (_Float16*)ws;
    size_t off = (size_t)N_PTS * D_DIM * sizeof(_Float16);
    float* rowPos    = (float*)(ws + off); off += (size_t)COL_CHUNKS * N_PTS * sizeof(float);
    float* rowPosCnt = (float*)(ws + off); off += (size_t)COL_CHUNKS * N_PTS * sizeof(float);
    float* rowNeg    = (float*)(ws + off); off += (size_t)COL_CHUNKS * N_PTS * sizeof(float);
    float* rowNegCnt = (float*)(ws + off); off += (size_t)COL_CHUNKS * N_PTS * sizeof(float);
    float* lastAcc   = (float*)(ws + off);

    const int nelem = N_PTS * D_DIM;
    convert_f32_to_f16<<<(nelem + 255) / 256, 256, 0, stream>>>(x, xh, nelem);
    gemm_loss<<<128, 256, 0, stream>>>(xh, tgt, rowPos, rowPosCnt, rowNeg, rowNegCnt, lastAcc);
    finalize<<<1, 256, 0, stream>>>(rowPos, rowPosCnt, rowNeg, rowNegCnt, lastAcc, out);
}
